// MultiHeadAttention_5162550690632
// MI455X (gfx1250) — matrix operand order repaired and checked by reading, not compiled, not run on hardware
//
#include <hip/hip_runtime.h>
#include <hip/hip_bf16.h>
#include <cstdint>
#include <cstddef>

// Problem constants (match reference)
#define B_    2
#define S_    2048
#define EMB_  1024
#define H_    16
#define D_    64
#define ND_   (H_ * D_)   // 1024

typedef __attribute__((ext_vector_type(16))) __bf16 v16bf;
typedef __attribute__((ext_vector_type(8)))  float  v8f;
typedef unsigned int u32x4 __attribute__((ext_vector_type(4)));
typedef int          i32x4 __attribute__((ext_vector_type(4)));
typedef int          i32x8 __attribute__((ext_vector_type(8)));

// Load a 16-element bf16 WMMA fragment as two 16-byte chunks (ds_load_b128 /
// global_load_b128). Callers guarantee 16-byte alignment.
__device__ __forceinline__ v16bf ld_frag(const __bf16* p0, const __bf16* p1) {
  union { uint4 u[2]; v16bf v; } t;
  t.u[0] = *(const uint4*)p0;
  t.u[1] = *(const uint4*)p1;
  return t.v;
}

// Convert 16 floats -> bf16 and store as two 16-byte chunks.
__device__ __forceinline__ void st16bf(__bf16* dst, const float* s) {
  union { uint4 u[2]; __bf16 b[16]; } t;
#pragma unroll
  for (int j = 0; j < 16; ++j) t.b[j] = (__bf16)s[j];
  ((uint4*)dst)[0] = t.u[0];
  ((uint4*)dst)[1] = t.u[1];
}

// Issue a 2D TDM load: global tile -> LDS. Descriptor per CDNA5 ISA §8.3/8.4.
// group0: w0 = count=1 (valid user D#), w1 = lds byte offset,
//         w2 = global_addr[31:0], w3 = global_addr[56:32] | type=2<<30.
// This toolchain uses the 6-arg builtin form (extra i32x8 group before cpol).
__device__ __forceinline__ void tdm_load_2d(uint32_t ldsOff, const void* gptr,
                                            i32x8 g1) {
  const uint64_t ga = (uint64_t)(uintptr_t)gptr;
  u32x4 g0;
  g0.x = 1u;
  g0.y = ldsOff;
  g0.z = (uint32_t)ga;
  g0.w = (uint32_t)((ga >> 32) & 0x01FFFFFFull) | (2u << 30);
  i32x4 z4 = {0, 0, 0, 0};
  i32x8 z8 = {0, 0, 0, 0, 0, 0, 0, 0};
  __builtin_amdgcn_tensor_load_to_lds(g0, g1, z4, z4, z8, 0);
}

// Low 32 bits of a generic pointer to LDS == LDS byte offset (flat LDS
// aperture mapping truncates to addr[31:0]).
__device__ __forceinline__ uint32_t lds_off(const void* p) {
  return (uint32_t)(uintptr_t)p;
}

// ---------------------------------------------------------------------------
// Kernel 1: C = X @ W^T + bias, output scattered to bf16 [B][H][S][64].
// Block = 256 thr = 8 waves; tile 128(M) x 128(N), K-step 32.
// Wave grid 4(M) x 2(N): each wave -> 32x64 = 2x4 WMMA accumulators.
// ---------------------------------------------------------------------------
__global__ __launch_bounds__(256)
void qkv_proj_kernel(const float* __restrict__ X, const float* __restrict__ W,
                     const float* __restrict__ bias, __bf16* __restrict__ dst) {
  __shared__ __bf16 As[128][40];   // [m][k], stride 80 B (16B-aligned rows)
  __shared__ __bf16 Bs[128][40];   // [n][k]

  const int tid  = threadIdx.x;
  const int lane = tid & 31;
  const int wid  = tid >> 5;
  const int r    = lane & 15;
  const int hf   = lane >> 4;
  const int m0   = blockIdx.x * 128;
  const int n0   = blockIdx.y * 128;
  const int wm   = (wid & 3) * 32;
  const int wn   = (wid >> 2) * 64;

  v8f acc[2][4];
#pragma unroll
  for (int i = 0; i < 2; ++i)
#pragma unroll
    for (int j = 0; j < 4; ++j)
#pragma unroll
      for (int e = 0; e < 8; ++e) acc[i][j][e] = 0.0f;

  const int lrow = tid >> 1;
  const int lk   = (tid & 1) * 16;

  for (int k0 = 0; k0 < EMB_; k0 += 32) {
    {
      float tmp[16];
      const float* srcA = X + (size_t)(m0 + lrow) * EMB_ + k0 + lk;
#pragma unroll
      for (int j = 0; j < 4; ++j) {
        float4 f = ((const float4*)srcA)[j];
        tmp[4*j] = f.x; tmp[4*j+1] = f.y; tmp[4*j+2] = f.z; tmp[4*j+3] = f.w;
      }
      st16bf(&As[lrow][lk], tmp);
      const float* srcB = W + (size_t)(n0 + lrow) * EMB_ + k0 + lk;
#pragma unroll
      for (int j = 0; j < 4; ++j) {
        float4 f = ((const float4*)srcB)[j];
        tmp[4*j] = f.x; tmp[4*j+1] = f.y; tmp[4*j+2] = f.z; tmp[4*j+3] = f.w;
      }
      st16bf(&Bs[lrow][lk], tmp);
    }
    __syncthreads();

    v16bf a[2], b[4];
#pragma unroll
    for (int mi = 0; mi < 2; ++mi) {
      const __bf16* p = &As[wm + mi * 16 + r][hf * 8];
      a[mi] = ld_frag(p, p + 16);
    }
#pragma unroll
    for (int ni = 0; ni < 4; ++ni) {
      const __bf16* p = &Bs[wn + ni * 16 + r][hf * 8];
      b[ni] = ld_frag(p, p + 16);
    }
#pragma unroll
    for (int mi = 0; mi < 2; ++mi)
#pragma unroll
      for (int ni = 0; ni < 4; ++ni)
        acc[mi][ni] = __builtin_amdgcn_wmma_f32_16x16x32_bf16(
            false, a[mi], false, b[ni], (short)0, acc[mi][ni], false, false);
    __syncthreads();
  }

#pragma unroll
  for (int mi = 0; mi < 2; ++mi) {
#pragma unroll
    for (int ni = 0; ni < 4; ++ni) {
      const int n  = n0 + wn + ni * 16 + r;
      const float bn = bias[n];
      const int h = n >> 6, d = n & 63;
#pragma unroll
      for (int vr = 0; vr < 8; ++vr) {
        const int m  = m0 + wm + mi * 16 + vr + 8 * hf;
        const int bb = m >> 11;
        const int ss = m & (S_ - 1);
        dst[(((size_t)bb * H_ + h) * S_ + ss) * D_ + d] = (__bf16)(acc[mi][ni][vr] + bn);
      }
    }
  }
}

// ---------------------------------------------------------------------------
// Kernel 2: flash attention, TDM-pipelined. Grid (S/128, H, B), 8 waves.
// Double-buffered LDS: K chunks + mask tiles arrive by Tensor Data Mover
// (issued by wave 0, synced by s_wait_tensorcnt + barrier); V slices are
// register-prefetched and committed transposed each iteration.
// ---------------------------------------------------------------------------
__global__ __launch_bounds__(256)
void attn_kernel(const __bf16* __restrict__ Qb, const __bf16* __restrict__ Kb,
                 const __bf16* __restrict__ Vb, const uint8_t* __restrict__ mask,
                 float* __restrict__ out) {
  __shared__ __bf16  Ks[2][32][72];    // TDM dest; 144B rows via TDM pad
  __shared__ uint8_t Ms[2][128][32];   // TDM dest; mask tile, packed 32B rows
  __shared__ __bf16  Vt[2][64][40];    // [d][key], manual transpose, 80B rows
  __shared__ __bf16  Ps[8][16][40];    // per-wave P tile (C->A relayout)

  const int tid  = threadIdx.x;
  const int lane = tid & 31;
  const int wid  = tid >> 5;
  const int r    = lane & 15;
  const int hf   = lane >> 4;
  const int bb   = blockIdx.z, hh = blockIdx.y;
  const int qblk  = blockIdx.x * 128;
  const int qbase = qblk + wid * 16;

  const __bf16* Qp = Qb + ((size_t)bb * H_ + hh) * S_ * D_;
  const __bf16* Kp = Kb + ((size_t)bb * H_ + hh) * S_ * D_;
  const __bf16* Vp = Vb + ((size_t)bb * H_ + hh) * S_ * D_;
  const uint8_t* Mp = mask + (size_t)bb * S_ * S_ + (size_t)qblk * S_;

  // ---- TDM descriptor group-1 constants ----
  // K tile: tile 64(d) x 32(keys) bf16 (data_size=1), tensor_dim0=64,
  // dim0_stride=64, pad_enable, pad every 32 dwords (code 4) by 4 dwords
  // (code 3) -> LDS row stride 144 B.
  i32x8 g1k;
  g1k[0] = (int)((1u << 16) | (1u << 20) | (4u << 22) | (3u << 25));
  g1k[1] = (int)(64u << 16);            // tensor_dim0 = 64
  g1k[2] = (int)((unsigned)S_ << 16);   // tensor_dim1 = 2048
  g1k[3] = (int)(64u << 16);            // tile_dim0 = 64
  g1k[4] = 32;                          // tile_dim1 = 32
  g1k[5] = 64;                          // tensor_dim0_stride = 64
  g1k[6] = 0; g1k[7] = 0;
  // mask tile: 32(k) x 128(q) bytes (data_size=0), row stride S_.
  i32x8 g1m;
  g1m[0] = 0;
  g1m[1] = (int)((unsigned)S_ << 16);   // tensor_dim0 = 2048
  g1m[2] = (int)((unsigned)S_ << 16);   // tensor_dim1 = 2048
  g1m[3] = (int)(32u << 16);            // tile_dim0 = 32 bytes
  g1m[4] = 128;                         // tile_dim1 = 128 rows
  g1m[5] = S_;                          // tensor_dim0_stride = 2048
  g1m[6] = 0; g1m[7] = 0;

  const uint32_t ksOff0 = lds_off(&Ks[0][0][0]), ksOff1 = lds_off(&Ks[1][0][0]);
  const uint32_t msOff0 = lds_off(&Ms[0][0][0]), msOff1 = lds_off(&Ms[1][0][0]);

  // ---- Q as WMMA A fragments (held in registers for the whole loop) ----
  v16bf aq[2];
  {
    const __bf16* qrow = Qp + (size_t)(qbase + r) * D_;
#pragma unroll
    for (int c = 0; c < 2; ++c) {
      const __bf16* p = qrow + c * 32 + hf * 8;
      aq[c] = ld_frag(p, p + 16);
    }
  }

  float m_i[8], l_i[8];
#pragma unroll
  for (int vr = 0; vr < 8; ++vr) { m_i[vr] = -__builtin_inff(); l_i[vr] = 0.0f; }
  v8f oacc[4];
#pragma unroll
  for (int ni = 0; ni < 4; ++ni)
#pragma unroll
    for (int e = 0; e < 8; ++e) oacc[ni][e] = 0.0f;

  const int srow = tid >> 3;        // 0..31 : key row this thread stages
  const int sd   = (tid & 7) * 8;   // 0..56 : 8 bf16 along d

  // ---- prologue: launch chunk 0 DMAs, prefetch V chunk 0 into registers ----
  if (wid == 0) {
    tdm_load_2d(ksOff0, Kp, g1k);
    tdm_load_2d(msOff0, Mp, g1m);
  }
  uint4 vreg = *(const uint4*)(Vp + (size_t)srow * D_ + sd);

  int buf = 0;
  for (int kb = 0; kb < S_; kb += 32) {
    // commit prefetched V slice (transposed) for the current chunk
    {
      union { uint4 u; __bf16 b[8]; } vv;
      vv.u = vreg;
#pragma unroll
      for (int j = 0; j < 8; ++j) Vt[buf][sd + j][srow] = vv.b[j];
    }
    if (wid == 0) __builtin_amdgcn_s_wait_tensorcnt(0);  // chunk's K + mask landed
    __syncthreads();                                     // publish K/mask/V

    // launch next chunk's DMAs / V register prefetch (into the other buffer)
    if (kb + 32 < S_) {
      if (wid == 0) {
        tdm_load_2d(buf ? ksOff0 : ksOff1, Kp + (size_t)(kb + 32) * D_, g1k);
        tdm_load_2d(buf ? msOff0 : msOff1, Mp + (kb + 32), g1m);
      }
      vreg = *(const uint4*)(Vp + (size_t)(kb + 32 + srow) * D_ + sd);
    }

    // ---- scores S = Q(16x64) x K^T(64x32): 2 key-subtiles x 2 d-chunks ----
    v8f sc[2];
#pragma unroll
    for (int t = 0; t < 2; ++t) {
#pragma unroll
      for (int e = 0; e < 8; ++e) sc[t][e] = 0.0f;
#pragma unroll
      for (int c = 0; c < 2; ++c) {
        const __bf16* p = &Ks[buf][t * 16 + r][c * 32 + hf * 8];
        v16bf bk = ld_frag(p, p + 16);
        sc[t] = __builtin_amdgcn_wmma_f32_16x16x32_bf16(
            false, aq[c], false, bk, (short)0, sc[t], false, false);
      }
    }

    // ---- scale + mask from the TDM-staged LDS tile ----
#pragma unroll
    for (int t = 0; t < 2; ++t) {
#pragma unroll
      for (int vr = 0; vr < 8; ++vr) {
        const uint8_t mv = Ms[buf][wid * 16 + vr + 8 * hf][t * 16 + r];
        const float s = sc[t][vr] * 0.125f;   // 1/sqrt(64)
        sc[t][vr] = mv ? -1e9f : s;           // reference: masked == exactly -1e9
      }
    }

    // ---- online softmax (rows live across a 16-lane half-wave) ----
#pragma unroll
    for (int vr = 0; vr < 8; ++vr) {
      float smax = fmaxf(sc[0][vr], sc[1][vr]);
#pragma unroll
      for (int xm = 1; xm < 16; xm <<= 1) smax = fmaxf(smax, __shfl_xor(smax, xm, 32));
      const float mnew  = fmaxf(m_i[vr], smax);
      const float alpha = __expf(m_i[vr] - mnew);
      const float p0 = __expf(sc[0][vr] - mnew);
      const float p1 = __expf(sc[1][vr] - mnew);
      float psum = p0 + p1;
#pragma unroll
      for (int xm = 1; xm < 16; xm <<= 1) psum += __shfl_xor(psum, xm, 32);
      l_i[vr] = l_i[vr] * alpha + psum;
      m_i[vr] = mnew;
#pragma unroll
      for (int ni = 0; ni < 4; ++ni) oacc[ni][vr] *= alpha;
      Ps[wid][vr + 8 * hf][r]      = (__bf16)p0;   // C layout -> LDS
      Ps[wid][vr + 8 * hf][16 + r] = (__bf16)p1;
    }
    __syncthreads();   // Ps relayout visible (and keeps ds order explicit)

    // ---- O += P(16x32) x V(32x64) ----
    v16bf ap;
    {
      const __bf16* p = &Ps[wid][r][hf * 8];
      ap = ld_frag(p, p + 16);
    }
#pragma unroll
    for (int ni = 0; ni < 4; ++ni) {
      const __bf16* p = &Vt[buf][ni * 16 + r][hf * 8];
      v16bf bv = ld_frag(p, p + 16);
      oacc[ni] = __builtin_amdgcn_wmma_f32_16x16x32_bf16(
          false, ap, false, bv, (short)0, oacc[ni], false, false);
    }
    buf ^= 1;
  }

  // ---- finalize: O / l, write f32 output [B][S][H*64] ----
#pragma unroll
  for (int ni = 0; ni < 4; ++ni) {
    const int d = ni * 16 + r;
#pragma unroll
    for (int vr = 0; vr < 8; ++vr) {
      const int q = qbase + vr + 8 * hf;
      out[((size_t)bb * S_ + q) * ND_ + hh * D_ + d] = oacc[ni][vr] / l_i[vr];
    }
  }
}

// ---------------------------------------------------------------------------
extern "C" void kernel_launch(void* const* d_in, const int* in_sizes, int n_in,
                              void* d_out, int out_size, void* d_ws, size_t ws_size,
                              hipStream_t stream) {
  (void)in_sizes; (void)n_in; (void)out_size; (void)ws_size;
  const float*   x_q  = (const float*)d_in[0];
  const float*   x_kv = (const float*)d_in[1];
  const uint8_t* mask = (const uint8_t*)d_in[2];
  const float*   w_q  = (const float*)d_in[3];
  const float*   b_q  = (const float*)d_in[4];
  const float*   w_k  = (const float*)d_in[5];
  const float*   b_k  = (const float*)d_in[6];
  const float*   w_v  = (const float*)d_in[7];
  const float*   b_v  = (const float*)d_in[8];
  float*         out  = (float*)d_out;

  // Workspace: bf16 Q, K, V each [B][H][S][64] = 8 MB -> 24 MB total.
  const size_t perBuf = (size_t)B_ * H_ * S_ * D_;
  __bf16* Qb = (__bf16*)d_ws;
  __bf16* Kb = Qb + perBuf;
  __bf16* Vb = Kb + perBuf;

  dim3 blk(256, 1, 1);
  dim3 g1((B_ * S_) / 128, ND_ / 128, 1);      // (32, 8)
  qkv_proj_kernel<<<g1, blk, 0, stream>>>(x_q,  w_q, b_q, Qb);
  qkv_proj_kernel<<<g1, blk, 0, stream>>>(x_kv, w_k, b_k, Kb);
  qkv_proj_kernel<<<g1, blk, 0, stream>>>(x_kv, w_v, b_v, Vb);

  dim3 g2(S_ / 128, H_, B_);                   // (16, 16, 2)
  attn_kernel<<<g2, blk, 0, stream>>>(Qb, Kb, Vb, mask, out);
}
